// SEMambapp_bottleneck_63342177681523
// MI455X (gfx1250) — matrix operand order repaired and checked by reading, not compiled, not run on hardware
//
#include <hip/hip_runtime.h>
#include <hip/hip_bf16.h>

// ---------------------------------------------------------------------------
// CDNA5 (gfx1250) implementation: all dense projections via
// v_wmma_f32_16x16x32_f16 with double-buffered LDS-staged f16 tiles
// (ds_load_b128 fragments, packed ds_store_b32 staging, 1 barrier/K-step).
// Block = 128 threads = 4 wave32 -> 64x64 C macro-tile, 4 WMMA tiles per wave.
// ---------------------------------------------------------------------------

typedef __attribute__((ext_vector_type(16))) _Float16 v16h;
typedef __attribute__((ext_vector_type(8)))  _Float16 v8h;
typedef __attribute__((ext_vector_type(2)))  _Float16 v2h;
typedef __attribute__((ext_vector_type(8)))  float    v8f;

#define T_DIM 200
#define LDS_PITCH 40   // halves per row (32 + 8 pad); 80B rows, 16B-aligned

__device__ __forceinline__ float silu_f(float x)     { return x / (1.f + __expf(-x)); }
__device__ __forceinline__ float gelu_f(float x)     { return 0.5f * x * (1.f + erff(x * 0.70710678118654752f)); }
__device__ __forceinline__ float softplus_f(float x) { return (x > 20.f) ? x : log1pf(__expf(x)); }

// pack two f32 -> packed f16 pair (v_cvt_pk_rtz_f16_f32)
__device__ __forceinline__ v2h pk2h(float a, float b) {
    return __builtin_bit_cast(v2h, __builtin_amdgcn_cvt_pkrtz(a, b));
}

// ---------------------------------------------------------------------------
// Strided WMMA GEMM:  Y[m,n] = epilogue( sum_k X[m,k] * W[n,k] )
//   X(m,k) at X[m*xrs + k*xcs]     W(n,k) at W[n*wns + k*wks]
//   Y(m,n) at Y[m*yrs + n*ycs]     R(m,n) at R[m*rrs + n*rcs] (optional)
//   act: 0=none 1=gelu 2=softplus   (bias added before act, R after)
// ---------------------------------------------------------------------------
__global__ void __launch_bounds__(128)
k_gemm_wmma(const float* __restrict__ X, const float* __restrict__ W,
            const float* __restrict__ bias, const float* __restrict__ R,
            float* __restrict__ Y,
            int M, int N, int K,
            long long xrs, long long xcs,
            long long wns, long long wks,
            long long yrs, long long ycs,
            long long rrs, long long rcs,
            int act)
{
    __shared__ __align__(16) _Float16 Xh[2][64 * LDS_PITCH];
    __shared__ __align__(16) _Float16 Wh[2][64 * LDS_PITCH];

    const int t    = threadIdx.x;
    const int lane = t & 31;
    const int wv   = t >> 5;
    const int row  = lane & 15;
    const int hi   = lane >> 4;            // 0 or 1 (wave32 half)
    const int m0   = blockIdx.y * 64;
    const int n0   = blockIdx.x * 64;

    // staging mapping: idx = t + 128*i over 1024 pair-slots;
    // r = idx>>4 (0..63), pair col c2 = (idx&15)*2
    // clamped unconditional loads + mask-multiply: no exec divergence.
    auto stage_regs = [&](int kk, v2h* sx, v2h* sw) {
#pragma unroll
        for (int i = 0; i < 8; ++i) {
            int idx = t + 128 * i;
            int r   = idx >> 4;
            int c2  = (idx & 15) << 1;
            int k0  = kk + c2;
            int k1  = k0 + 1;
            int kc0 = (k0 < K) ? k0 : (K - 1);
            int kc1 = (k1 < K) ? k1 : (K - 1);

            int m  = m0 + r;
            int mc = (m < M) ? m : (M - 1);
            const float* xb = X + (long long)mc * xrs;
            float x0 = xb[(long long)kc0 * xcs] * (((m < M) & (k0 < K)) ? 1.f : 0.f);
            float x1 = xb[(long long)kc1 * xcs] * (((m < M) & (k1 < K)) ? 1.f : 0.f);
            sx[i] = pk2h(x0, x1);

            int nn = n0 + r;
            int nc = (nn < N) ? nn : (N - 1);
            const float* wb = W + (long long)nc * wns;
            float w0 = wb[(long long)kc0 * wks] * (((nn < N) & (k0 < K)) ? 1.f : 0.f);
            float w1 = wb[(long long)kc1 * wks] * (((nn < N) & (k1 < K)) ? 1.f : 0.f);
            sw[i] = pk2h(w0, w1);
        }
    };
    auto store_lds = [&](const v2h* sx, const v2h* sw, _Float16* Xb, _Float16* Wb) {
#pragma unroll
        for (int i = 0; i < 8; ++i) {
            int idx = t + 128 * i;
            int r   = idx >> 4;
            int c2  = (idx & 15) << 1;
            *(v2h*)&Xb[r * LDS_PITCH + c2] = sx[i];   // ds_store_b32
            *(v2h*)&Wb[r * LDS_PITCH + c2] = sw[i];
        }
    };

    v8f acc[4] = {};
    v2h sx[8], sw[8];

    // prologue: stage first tile into buffer 0
    stage_regs(0, sx, sw);
    store_lds(sx, sw, Xh[0], Wh[0]);
    __syncthreads();

    int p = 0;
    for (int kk = 0; kk < K; kk += 32) {
        const bool more = (kk + 32) < K;
        // issue next tile's global loads early: latency hides behind WMMAs
        if (more) stage_regs(kk + 32, sx, sw);

        // ---- A fragment (16x32 f16 layout): two contiguous 8-half runs ----
        const _Float16* Xb = Xh[p];
        const _Float16* Wb = Wh[p];
        const int arow = (wv * 16 + row) * LDS_PITCH;
        v8h alo = *(const v8h*)&Xb[arow + (hi << 3)];
        v8h ahi = *(const v8h*)&Xb[arow + 16 + (hi << 3)];
        v16h a = __builtin_shufflevector(alo, ahi,
                                         0, 1, 2, 3, 4, 5, 6, 7,
                                         8, 9, 10, 11, 12, 13, 14, 15);

        // ---- 4 B fragments (32x16 f16 layout): two contiguous 8-half runs ----
#pragma unroll
        for (int j = 0; j < 4; ++j) {
            const int brow = (j * 16 + row) * LDS_PITCH + (hi << 3);
            v8h blo = *(const v8h*)&Wb[brow];
            v8h bhi = *(const v8h*)&Wb[brow + 16];
            v16h b = __builtin_shufflevector(blo, bhi,
                                             0, 1, 2, 3, 4, 5, 6, 7,
                                             8, 9, 10, 11, 12, 13, 14, 15);
            acc[j] = __builtin_amdgcn_wmma_f32_16x16x32_f16(false, a, false, b,
                                                            (short)0, acc[j],
                                                            false, false);
        }

        // write next tile to the other buffer; one barrier per K-step
        if (more) store_lds(sx, sw, Xh[p ^ 1], Wh[p ^ 1]);
        __syncthreads();
        p ^= 1;
    }

    // C/D 16x16 f32: lanes0-15 N=lane, VGPR r -> M=r; lanes16-31 N=lane-16, M=8+r
#pragma unroll
    for (int j = 0; j < 4; ++j) {
        const int n_out = n0 + j * 16 + row;
#pragma unroll
        for (int r = 0; r < 8; ++r) {
            int m_out = m0 + wv * 16 + (hi << 3) + r;
            if (m_out < M && n_out < N) {
                float v = acc[j][r];
                if (bias) v += bias[n_out];
                if (act == 1)      v = gelu_f(v);
                else if (act == 2) v = softplus_f(v);
                if (R) v += R[(long long)m_out * rrs + (long long)n_out * rcs];
                Y[(long long)m_out * yrs + (long long)n_out * ycs] = v;
            }
        }
    }
}

// ---------------------------------------------------------------------------
// Generic 3D gather-permute (handles transposes and flips via signed strides)
// out[(i0*D1+i1)*D2+i2] = in[base + i0*s0 + i1*s1 + i2*s2]
// ---------------------------------------------------------------------------
__global__ void k_perm3(const float* __restrict__ in, float* __restrict__ out,
                        int D1, int D2,
                        long long s0, long long s1, long long s2,
                        long long base, long long n)
{
    long long i = (long long)blockIdx.x * blockDim.x + threadIdx.x;
    if (i >= n) return;
    int i2 = (int)(i % D2);
    long long r = i / D2;
    int i1 = (int)(r % D1);
    long long i0 = r / D1;
    out[i] = in[base + i0 * s0 + (long long)i1 * s1 + (long long)i2 * s2];
}

// LayerNorm (mode 0, affine w,b) or RMSNorm (mode 1, w only) over C elements/row.
// Element (row,k) at x[row*xrs + k*xcs]; one wave32 per row.
__global__ void k_norm(const float* __restrict__ x, const float* __restrict__ w,
                       const float* __restrict__ b, float* __restrict__ y,
                       long long rows, int C,
                       long long xrs, long long xcs, long long yrs, long long ycs,
                       float eps, int mode)
{
    long long row = blockIdx.x;
    if (row >= rows) return;
    int lane = threadIdx.x;
    float s = 0.f, ss = 0.f;
    for (int k = lane; k < C; k += 32) {
        float v = x[row * xrs + (long long)k * xcs];
        s += v; ss += v * v;
    }
    for (int off = 16; off; off >>= 1) {
        s  += __shfl_xor(s, off);
        ss += __shfl_xor(ss, off);
    }
    float mu = 0.f, inv;
    if (mode == 0) { mu = s / C; inv = rsqrtf(ss / C - mu * mu + eps); }
    else           { inv = rsqrtf(ss / C + eps); }
    for (int k = lane; k < C; k += 32) {
        float v = (x[row * xrs + (long long)k * xcs] - mu) * inv * w[k];
        if (mode == 0) v += b[k];
        y[row * yrs + (long long)k * ycs] = v;
    }
}

// Depthwise causal conv (k=4) over L on the first `di` channels of xz rows, +bias, SiLU
__global__ void k_dwconv_silu(const float* __restrict__ xz, const float* __restrict__ w,
                              const float* __restrict__ bias, float* __restrict__ xc,
                              int Bn, int L, int di)
{
    long long i = (long long)blockIdx.x * blockDim.x + threadIdx.x;
    long long n = (long long)Bn * L * di;
    if (i >= n) return;
    int d = (int)(i % di);
    int l = (int)((i / di) % L);
    int bn = (int)(i / ((long long)di * L));
    float acc = bias[d];
#pragma unroll
    for (int j = 0; j < 4; ++j) {
        int lj = l - 3 + j;
        if (lj >= 0)
            acc += w[d * 4 + j] * xz[((long long)(bn * L + lj)) * (2 * di) + d];
    }
    xc[i] = silu_f(acc);
}

// Selective scan: one thread per (bn, d); DS=16 state in registers, L steps.
__global__ void k_scan(const float* __restrict__ dt, const float* __restrict__ xc,
                       const float* __restrict__ xdb, const float* __restrict__ A_log,
                       float* __restrict__ ys,
                       int L, int di, int nxdb, int dtr)
{
    int d = blockIdx.x * blockDim.x + threadIdx.x;
    int bn = blockIdx.y;
    if (d >= di) return;
    float A[16], h[16];
#pragma unroll
    for (int s = 0; s < 16; ++s) { A[s] = -__expf(A_log[d * 16 + s]); h[s] = 0.f; }
    for (int l = 0; l < L; ++l) {
        long long t = (long long)bn * L + l;
        float dtv = dt[t * di + d];
        float xv  = xc[t * di + d];
        const float* Bp = xdb + t * nxdb + dtr;
        const float* Cp = Bp + 16;
        float dx = dtv * xv, y = 0.f;
#pragma unroll
        for (int s = 0; s < 16; ++s) {
            h[s] = __expf(dtv * A[s]) * h[s] + dx * Bp[s];
            y += h[s] * Cp[s];
        }
        ys[t * di + d] = y;
    }
}

// y = (yscan + xc*D[d]) * silu(z) ; z = xz[..., di+d]
__global__ void k_ymul(const float* __restrict__ ys, const float* __restrict__ xc,
                       const float* __restrict__ xz, const float* __restrict__ Dp,
                       float* __restrict__ y, long long tok, int di)
{
    long long i = (long long)blockIdx.x * blockDim.x + threadIdx.x;
    long long n = tok * di;
    if (i >= n) return;
    int d = (int)(i % di);
    long long t = i / di;
    float z = xz[t * (2ll * di) + di + d];
    y[i] = (ys[i] + xc[i] * Dp[d]) * silu_f(z);
}

__global__ void k_cat2(const float* __restrict__ a, const float* __restrict__ b,
                       float* __restrict__ o, long long rows, int ca, int cb)
{
    long long i = (long long)blockIdx.x * blockDim.x + threadIdx.x;
    int cc = ca + cb;
    long long n = rows * cc;
    if (i >= n) return;
    int col = (int)(i % cc);
    long long r = i / cc;
    o[i] = (col < ca) ? a[r * ca + col] : b[r * cb + (col - ca)];
}

__global__ void k_mul(const float* __restrict__ a, const float* __restrict__ b,
                      float* __restrict__ o, long long n)
{
    long long i = (long long)blockIdx.x * blockDim.x + threadIdx.x;
    if (i < n) o[i] = a[i] * b[i];
}

__global__ void k_add(const float* __restrict__ a, const float* __restrict__ b,
                      float* __restrict__ o, long long n)
{
    long long i = (long long)blockIdx.x * blockDim.x + threadIdx.x;
    if (i < n) o[i] = a[i] + b[i];
}

// fan*g : out[m, n] = (n<c ? cos(pr[m,n]) : sin(pr[m,n-c])) * g[m,n]
__global__ void k_fan_gate(const float* __restrict__ pr, const float* __restrict__ g,
                           float* __restrict__ o, long long rows, int c)
{
    long long i = (long long)blockIdx.x * blockDim.x + threadIdx.x;
    long long n = rows * 2ll * c;
    if (i >= n) return;
    int col = (int)(i % (2 * c));
    long long r = i / (2 * c);
    float p = (col < c) ? cosf(pr[r * c + col]) : sinf(pr[r * c + (col - c)]);
    o[i] = p * g[i];
}

// Conv2d (3,4), stride (1,2), pad (1,1), NCHW
__global__ void k_conv_down(const float* __restrict__ x, const float* __restrict__ w,
                            const float* __restrict__ b, float* __restrict__ y,
                            int ci, int co, int T, int Fi, int Fo)
{
    long long i = (long long)blockIdx.x * blockDim.x + threadIdx.x;
    long long n = (long long)co * T * Fo;
    if (i >= n) return;
    int wo = (int)(i % Fo);
    int th = (int)((i / Fo) % T);
    int o  = (int)(i / ((long long)Fo * T));
    float acc = b[o];
    for (int ii = 0; ii < ci; ++ii)
        for (int kh = 0; kh < 3; ++kh) {
            int h = th + kh - 1;
            if (h < 0 || h >= T) continue;
            for (int kw = 0; kw < 4; ++kw) {
                int wc = wo * 2 + kw - 1;
                if (wc < 0 || wc >= Fi) continue;
                acc += x[((long long)ii * T + h) * Fi + wc] *
                       w[((o * ci + ii) * 3 + kh) * 4 + kw];
            }
        }
    y[i] = acc;
}

// ConvTranspose2d (3,4), stride (1,2), pad (1,1), weight (ci,co,3,4), gather form
__global__ void k_conv_up(const float* __restrict__ x, const float* __restrict__ w,
                          const float* __restrict__ b, float* __restrict__ y,
                          int ci, int co, int T, int Fi, int Fo)
{
    long long i = (long long)blockIdx.x * blockDim.x + threadIdx.x;
    long long n = (long long)co * T * Fo;
    if (i >= n) return;
    int wo = (int)(i % Fo);
    int th = (int)((i / Fo) % T);
    int o  = (int)(i / ((long long)Fo * T));
    float acc = b[o];
    for (int ii = 0; ii < ci; ++ii)
        for (int kh = 0; kh < 3; ++kh) {
            int hi = th + 1 - kh;
            if (hi < 0 || hi >= T) continue;
            for (int kw = 0; kw < 4; ++kw) {
                int tw = wo + 1 - kw;
                if (tw < 0 || (tw & 1)) continue;
                int wi = tw >> 1;
                if (wi >= Fi) continue;
                acc += x[((long long)ii * T + hi) * Fi + wi] *
                       w[((ii * co + o) * 3 + kh) * 4 + kw];
            }
        }
    y[i] = acc;
}

// Per-channel mean / rsqrt(var+eps) over HW (InstanceNorm stats)
__global__ void k_chan_stats(const float* __restrict__ x, float* __restrict__ st,
                             int HW)
{
    int c = blockIdx.x;
    __shared__ float sm[256], sm2[256];
    float s = 0.f, ss = 0.f;
    for (int i = threadIdx.x; i < HW; i += 256) {
        float v = x[(long long)c * HW + i];
        s += v; ss += v * v;
    }
    sm[threadIdx.x] = s; sm2[threadIdx.x] = ss;
    __syncthreads();
    for (int o = 128; o; o >>= 1) {
        if (threadIdx.x < o) { sm[threadIdx.x] += sm[threadIdx.x + o];
                               sm2[threadIdx.x] += sm2[threadIdx.x + o]; }
        __syncthreads();
    }
    if (threadIdx.x == 0) {
        float mu = sm[0] / HW;
        st[2 * c]     = mu;
        st[2 * c + 1] = rsqrtf(sm2[0] / HW - mu * mu + 1e-5f);
    }
}

__global__ void k_in_prelu(const float* __restrict__ x, const float* __restrict__ st,
                           const float* __restrict__ w, const float* __restrict__ bb,
                           const float* __restrict__ a, float* __restrict__ y,
                           int C, int HW)
{
    long long i = (long long)blockIdx.x * blockDim.x + threadIdx.x;
    long long n = (long long)C * HW;
    if (i >= n) return;
    int c = (int)(i / HW);
    float v = (x[i] - st[2 * c]) * st[2 * c + 1] * w[c] + bb[c];
    y[i] = (v > 0.f) ? v : a[c] * v;
}

// ---------------------------------------------------------------------------
// Host side
// ---------------------------------------------------------------------------
struct MambaP { const float *norm_w,*in_w,*conv_w,*conv_b,*xp_w,*dt_w,*dt_b,*A_log,*D,*out_w; };
struct ConvP  { const float *w,*b,*in_w,*in_b,*a; };
struct GateP  { const float *w,*b,*ln_w,*ln_b; };
struct LevelP {
    MambaP fwd, bwd;
    const float *tl_w,*tl_b,*ln_w,*ln_b,*f_w1,*f_b1,*f_w2,*f_b2,*f_w3,*f_b3,
                *c_wp,*c_wg,*c_bg,*c_wo,*c_bo;
};

extern "C" void kernel_launch(void* const* d_in, const int* in_sizes, int n_in,
                              void* d_out, int out_size, void* d_ws, size_t ws_size,
                              hipStream_t stream)
{
    (void)in_sizes; (void)n_in; (void)out_size; (void)ws_size;
    int idx = 0;
    auto nxt = [&]() { return (const float*)d_in[idx++]; };

    const float* X0 = nxt();                           // x (1,64,200,100)
    ConvP down[2];
    for (int i = 0; i < 2; ++i) down[i] = { nxt(), nxt(), nxt(), nxt(), nxt() };
    LevelP lv[3];
    auto readM = [&](MambaP& m) {
        m.norm_w=nxt(); m.in_w=nxt(); m.conv_w=nxt(); m.conv_b=nxt(); m.xp_w=nxt();
        m.dt_w=nxt(); m.dt_b=nxt(); m.A_log=nxt(); m.D=nxt(); m.out_w=nxt();
    };
    for (int i = 0; i < 3; ++i) {
        readM(lv[i].fwd); readM(lv[i].bwd);
        lv[i].tl_w=nxt(); lv[i].tl_b=nxt(); lv[i].ln_w=nxt(); lv[i].ln_b=nxt();
        lv[i].f_w1=nxt(); lv[i].f_b1=nxt(); lv[i].f_w2=nxt(); lv[i].f_b2=nxt();
        lv[i].f_w3=nxt(); lv[i].f_b3=nxt();
        lv[i].c_wp=nxt(); lv[i].c_wg=nxt(); lv[i].c_bg=nxt(); lv[i].c_wo=nxt(); lv[i].c_bo=nxt();
    }
    ConvP up[2];
    for (int i = 0; i < 2; ++i) up[i] = { nxt(), nxt(), nxt(), nxt(), nxt() };
    GateP gt[2];
    for (int i = 0; i < 2; ++i) gt[i] = { nxt(), nxt(), nxt(), nxt() };

    const int cL[3] = {64, 128, 256};
    const int fL[3] = {100, 50, 25};

    // ---- workspace arena (floats); c*T*f == 1,280,000 at every level ----
    const long long S1 = 1280000;
    float* wsf = (float*)d_ws;
    long long off = 0;
    auto al = [&](long long nf) { float* p = wsf + off; off += nf; return p; };
    float* p_x2  = al(S1);  float* p_x3  = al(S1);
    float* p_y1  = al(S1);  float* p_y2  = al(S1);  float* p_y3 = al(S1);
    float* p_u2  = al(S1);  float* p_g2  = al(S1);
    float* p_u1  = al(S1);  float* p_g1  = al(S1);
    float* p_cat = al(2*S1);
    float* p_st  = al(1024);
    float* t_xs  = al(S1);  float* t_rms = al(S1);
    float* t_xz  = al(8*S1); float* t_xc = al(4*S1);
    float* t_xdb = al(S1);  float* t_dt  = al(4*S1);
    float* t_ys  = al(4*S1);
    float* t_y   = t_dt;                     // alias: dt dead after scan
    float* t_yf  = al(S1);  float* t_yd = al(S1); float* t_yb = al(S1);
    float* t_xr  = al(S1);  float* t_m  = al(2*S1);
    float* t_xs2 = al(S1);  float* t_x2 = al(S1); float* t_x3 = al(S1);
    float* t_h1  = al(2*S1); float* t_h2 = al(2*S1);
    float* t_x3b = al(S1);  float* t_pr = al(S1);
    float* t_g   = al(2*S1); float* t_fg = al(2*S1);

    auto nb = [](long long n) { return dim3((unsigned)((n + 255) / 256)); };
    auto gemm = [&](const float* A, const float* W, const float* bias, const float* R,
                    float* Y, int M, int N, int K,
                    long long xrs, long long xcs, long long wns, long long wks,
                    long long yrs, long long ycs, long long rrs, long long rcs, int act) {
        dim3 g((N + 63) / 64, (M + 63) / 64);
        k_gemm_wmma<<<g, 128, 0, stream>>>(A, W, bias, R, Y, M, N, K,
                                           xrs, xcs, wns, wks, yrs, ycs, rrs, rcs, act);
    };
    auto perm3 = [&](const float* in, float* out, int D0, int D1, int D2,
                     long long s0, long long s1, long long s2, long long base) {
        long long n = (long long)D0 * D1 * D2;
        k_perm3<<<nb(n), 256, 0, stream>>>(in, out, D1, D2, s0, s1, s2, base, n);
    };

    // x: (tok, c) contiguous (direction order handled by caller). yout = mamba(rms(x)) + x
    auto run_mamba = [&](const float* xin, const MambaP& P, int c, int f, float* yout) {
        int di = 4 * c, dtr = c / 16, tok = f * T_DIM, nxdb = dtr + 32;
        k_norm<<<(unsigned)tok, 32, 0, stream>>>(xin, P.norm_w, nullptr, t_rms,
                                                 tok, c, c, 1, c, 1, 1e-5f, 1);
        gemm(t_rms, P.in_w, nullptr, nullptr, t_xz, tok, 2 * di, c,
             c, 1, c, 1, 2 * di, 1, 0, 0, 0);
        k_dwconv_silu<<<nb((long long)tok * di), 256, 0, stream>>>(t_xz, P.conv_w, P.conv_b,
                                                                   t_xc, f, T_DIM, di);
        gemm(t_xc, P.xp_w, nullptr, nullptr, t_xdb, tok, nxdb, di,
             di, 1, di, 1, nxdb, 1, 0, 0, 0);
        gemm(t_xdb, P.dt_w, P.dt_b, nullptr, t_dt, tok, di, dtr,
             nxdb, 1, dtr, 1, di, 1, 0, 0, 2);
        k_scan<<<dim3((di + 255) / 256, f), 256, 0, stream>>>(t_dt, t_xc, t_xdb, P.A_log,
                                                              t_ys, T_DIM, di, nxdb, dtr);
        k_ymul<<<nb((long long)tok * di), 256, 0, stream>>>(t_ys, t_xc, t_xz, P.D,
                                                            t_y, tok, di);
        gemm(t_y, P.out_w, nullptr, xin, yout, tok, c, di,
             di, 1, di, 1, c, 1, c, 1, 0);
    };

    auto run_level = [&](const float* xl, const LevelP& L, int c, int f, float* yout) {
        long long tok = (long long)f * T_DIM;
        // xs (f,t,c) <- xl (c,t,f)
        perm3(xl, t_xs, f, T_DIM, c, 1, f, (long long)T_DIM * f, 0);
        run_mamba(t_xs, L.fwd, c, f, t_yf);
        // flip along t
        perm3(t_xs, t_xr, f, T_DIM, c, (long long)T_DIM * c, -(long long)c, 1,
              (long long)(T_DIM - 1) * c);
        run_mamba(t_xr, L.bwd, c, f, t_yd);
        perm3(t_yd, t_yb, f, T_DIM, c, (long long)T_DIM * c, -(long long)c, 1,
              (long long)(T_DIM - 1) * c);
        k_cat2<<<nb(tok * 2 * c), 256, 0, stream>>>(t_yf, t_yb, t_m, tok, c, c);
        // tlinear + residual
        gemm(t_m, L.tl_w, L.tl_b, t_xs, t_xs2, (int)tok, c, 2 * c,
             2 * c, 1, 1, c, c, 1, c, 1, 0);
        // (t,f,c) <- (f,t,c); LN over c; (c,t,f)
        perm3(t_xs2, t_x2, T_DIM, f, c, c, (long long)T_DIM * c, 1, 0);
        k_norm<<<(unsigned)tok, 32, 0, stream>>>(t_x2, L.ln_w, L.ln_b, t_x2,
                                                 tok, c, c, 1, c, 1, 1e-5f, 0);
        perm3(t_x2, t_x3, c, T_DIM, f, 1, (long long)f * c, c, 0);
        // frequency GLU (rows = c*T, K over f)
        int M2 = c * T_DIM;
        gemm(t_x3, L.f_w1, L.f_b1, nullptr, t_h1, M2, 2 * f, f,
             f, 1, 1, 2 * f, 2 * f, 1, 0, 0, 1);
        gemm(t_x3, L.f_w2, L.f_b2, nullptr, t_h2, M2, 2 * f, f,
             f, 1, 1, 2 * f, 2 * f, 1, 0, 0, 0);
        k_mul<<<nb((long long)M2 * 2 * f), 256, 0, stream>>>(t_h1, t_h2, t_h1,
                                                             (long long)M2 * 2 * f);
        gemm(t_h1, L.f_w3, L.f_b3, t_x3, t_x3b, M2, f, 2 * f,
             2 * f, 1, 1, f, f, 1, f, 1, 0);
        // FAN channel FFN (rows = t*f, K over c, X strided through (c,t,f))
        gemm(t_x3b, L.c_wp, nullptr, nullptr, t_pr, (int)tok, c, c,
             1, tok, 1, c, c, 1, 0, 0, 0);
        gemm(t_x3b, L.c_wg, L.c_bg, nullptr, t_g, (int)tok, 2 * c, c,
             1, tok, 1, 2 * c, 2 * c, 1, 0, 0, 1);
        k_fan_gate<<<nb(tok * 2 * c), 256, 0, stream>>>(t_pr, t_g, t_fg, tok, c);
        gemm(t_fg, L.c_wo, L.c_bo, t_x3b, yout, (int)tok, c, 2 * c,
             2 * c, 1, 1, c, 1, tok, 1, tok, 0);
    };

    auto run_inprelu = [&](float* buf, const ConvP& P, int C, int HW) {
        k_chan_stats<<<(unsigned)C, 256, 0, stream>>>(buf, p_st, HW);
        k_in_prelu<<<nb((long long)C * HW), 256, 0, stream>>>(buf, p_st, P.in_w, P.in_b,
                                                              P.a, buf, C, HW);
    };

    // ----------------- forward graph -----------------
    k_conv_down<<<nb((long long)128 * T_DIM * 50), 256, 0, stream>>>(X0, down[0].w, down[0].b,
                                                                     p_x2, 64, 128, T_DIM, 100, 50);
    run_inprelu(p_x2, down[0], 128, T_DIM * 50);
    k_conv_down<<<nb((long long)256 * T_DIM * 25), 256, 0, stream>>>(p_x2, down[1].w, down[1].b,
                                                                     p_x3, 128, 256, T_DIM, 50, 25);
    run_inprelu(p_x3, down[1], 256, T_DIM * 25);

    run_level(X0,   lv[0], cL[0], fL[0], p_y1);
    run_level(p_x2, lv[1], cL[1], fL[1], p_y2);
    run_level(p_x3, lv[2], cL[2], fL[2], p_y3);

    // u2 = up(y3)
    k_conv_up<<<nb((long long)128 * T_DIM * 50), 256, 0, stream>>>(p_y3, up[0].w, up[0].b,
                                                                   p_u2, 256, 128, T_DIM, 25, 50);
    run_inprelu(p_u2, up[0], 128, T_DIM * 50);
    // g2 = gate(cat(y2,u2))
    (void)hipMemcpyAsync(p_cat,      p_y2, S1 * sizeof(float), hipMemcpyDeviceToDevice, stream);
    (void)hipMemcpyAsync(p_cat + S1, p_u2, S1 * sizeof(float), hipMemcpyDeviceToDevice, stream);
    {
        int HW = T_DIM * 50;
        gemm(p_cat, gt[0].w, gt[0].b, nullptr, t_xs, HW, 128, 256,
             1, HW, 256, 1, 1, HW, 0, 0, 1);
        k_norm<<<(unsigned)HW, 32, 0, stream>>>(t_xs, gt[0].ln_w, gt[0].ln_b, p_g2,
                                                HW, 128, 1, HW, 1, HW, 1e-6f, 0);
    }
    // u1 = up(g2)
    k_conv_up<<<nb((long long)64 * T_DIM * 100), 256, 0, stream>>>(p_g2, up[1].w, up[1].b,
                                                                   p_u1, 128, 64, T_DIM, 50, 100);
    run_inprelu(p_u1, up[1], 64, T_DIM * 100);
    // g1 = gate(cat(y1,u1))
    (void)hipMemcpyAsync(p_cat,      p_y1, S1 * sizeof(float), hipMemcpyDeviceToDevice, stream);
    (void)hipMemcpyAsync(p_cat + S1, p_u1, S1 * sizeof(float), hipMemcpyDeviceToDevice, stream);
    {
        int HW = T_DIM * 100;
        gemm(p_cat, gt[1].w, gt[1].b, nullptr, t_xs, HW, 64, 128,
             1, HW, 128, 1, 1, HW, 0, 0, 1);
        k_norm<<<(unsigned)HW, 32, 0, stream>>>(t_xs, gt[1].ln_w, gt[1].ln_b, p_g1,
                                                HW, 64, 1, HW, 1, HW, 1e-6f, 0);
    }
    // out = y1 + g1
    k_add<<<nb(S1), 256, 0, stream>>>(p_y1, p_g1, (float*)d_out, S1);
}
